// VSSMBlock_12086037971084
// MI455X (gfx1250) — compile-verified, matrix operand order checked by reading, not run on hardware
//
#include <hip/hip_runtime.h>
#include <math.h>

typedef __attribute__((ext_vector_type(16))) _Float16 v16h;
typedef __attribute__((ext_vector_type(8)))  _Float16 v8h;
typedef __attribute__((ext_vector_type(8)))  float    v8f;

#define BATCH 2
#define CDIM 96
#define HH 64
#define WW 64
#define LL (HH*WW)          // 4096
#define DINN 192
#define KD 4
#define NST 16
#define RTK 6
#define XPR 38              // RTK + 2*NST
#define XPR_MPAD 48
#define DT_KPAD 32

__device__ __forceinline__ float siluf(float v){ return v * (1.f / (1.f + __expf(-v))); }
__device__ __forceinline__ float softplusf(float v){ return (v > 20.f) ? v : log1pf(__expf(v)); }
__device__ __forceinline__ float geluf(float v){ return 0.5f * v * (1.f + erff(v * 0.70710678118654752f)); }

union frag16 { v16h v; v8h h[2]; };

// ------------------------------------------------------------------
// Pack f32 weights (nK, Msrc, Ksrc) -> f16 (nK, Mpad, Kpad) zero-padded
// ------------------------------------------------------------------
__global__ void pack_w16(const float* __restrict__ src, _Float16* __restrict__ dst,
                         int Msrc, int Ksrc, int Mpad, int Kpad, int nK)
{
  int idx = blockIdx.x * 256 + threadIdx.x;
  int tot = nK * Mpad * Kpad;
  if (idx >= tot) return;
  int kk = idx % Kpad;
  int m  = (idx / Kpad) % Mpad;
  int k  = idx / (Kpad * Mpad);
  float v = (m < Msrc && kk < Ksrc) ? src[((long)k * Msrc + m) * Ksrc + kk] : 0.f;
  dst[idx] = (_Float16)v;
}

// ------------------------------------------------------------------
// LayerNorm2d over Cn channels (channel-planar (B,C,L)), write f16
// ------------------------------------------------------------------
__global__ void ln2d_f16(const float* __restrict__ x, const float* __restrict__ w,
                         const float* __restrict__ b, _Float16* __restrict__ out, int Cn)
{
  int p = blockIdx.x * 256 + threadIdx.x;           // pixel over BATCH*LL
  if (p >= BATCH * LL) return;
  int l = p % LL, bb = p / LL;
  const float* base = x + (long)bb * Cn * LL + l;
  float s = 0.f, s2 = 0.f;
  for (int c = 0; c < Cn; ++c) { float v = base[(long)c * LL]; s += v; s2 += v * v; }
  float mu = s / (float)Cn;
  float rs = rsqrtf(s2 / (float)Cn - mu * mu + 1e-5f);
  _Float16* ob = out + (long)bb * Cn * LL + l;
  for (int c = 0; c < Cn; ++c) {
    float v = (base[(long)c * LL] - mu) * rs * w[c] + b[c];
    ob[(long)c * LL] = (_Float16)v;
  }
}

// ------------------------------------------------------------------
// WMMA GEMM, block = 8 waves sharing one 16-column strip.
//   A (weights, f16 MxKP row-major): per lane 2x global b128 loads
//     (ISA 7.12.2: lane's 16 halves = K runs [kb..kb+7], [16+kb..16+kb+7])
//   B (activations, f16 KPxN channel-planar): 32x16 subtile staged
//     *transposed* into LDS once per k-step by all 256 threads, then
//     each lane reads 2x ds_load_b128. Bank-padded stride 40 halves.
//   K-loop fully unrolled via template KP.
// ------------------------------------------------------------------
template<int KP>
__global__ void __launch_bounds__(256) wmma_gemm(
    const _Float16* __restrict__ A, int aStrideK,
    const _Float16* __restrict__ Bm, long bStride,
    float* __restrict__ C, long cStride,
    _Float16* __restrict__ C16, long c16Stride,
    const float* __restrict__ bias, int biasStrideK,
    const float* __restrict__ res, long resStride,
    int M, int Mact, int N, int nK, int act, int mgroups)
{
  const int wave = threadIdx.x >> 5;
  const int lane = threadIdx.x & 31;
  const int s = blockIdx.y;                 // slice = b*nK + k
  const int k = s % nK;
  const _Float16* Ak = A + (long)k * aStrideK;
  const _Float16* Bs = Bm + (long)s * bStride;
  const int mg = blockIdx.x % mgroups;
  const int N0 = (blockIdx.x / mgroups) << 4;
  const int M0 = (mg << 7) + (wave << 4);
  const bool active = (M0 < M);             // wave-uniform (EXEC stays full)
  const int r16 = lane & 15;
  const int kb  = (lane < 16) ? 0 : 8;

  __shared__ _Float16 sB[16][40];           // [n][kk] transposed, padded

  const int skk = threadIdx.x >> 3;         // 0..31 : K row staged
  const int snp = (threadIdx.x & 7) << 1;   // 0,2,..14 : N pair staged

  v8f acc = {};
  #pragma unroll
  for (int k0 = 0; k0 < KP; k0 += 32) {
    {
      const _Float16* g = Bs + (long)(k0 + skk) * N + N0 + snp;
      unsigned u = *(const unsigned*)g;     // two halves n=snp, snp+1
      sB[snp][skk]     = ((const _Float16*)&u)[0];
      sB[snp + 1][skk] = ((const _Float16*)&u)[1];
    }
    __syncthreads();
    if (active) {
      const _Float16* arow = Ak + (long)(M0 + r16) * KP + k0;
      frag16 a, b;
      a.h[0] = *(const v8h*)(arow + kb);
      a.h[1] = *(const v8h*)(arow + 16 + kb);
      b.h[0] = *(const v8h*)&sB[r16][kb];
      b.h[1] = *(const v8h*)&sB[r16][16 + kb];
      acc = __builtin_amdgcn_wmma_f32_16x16x32_f16(false, a.v, false, b.v,
                                                   (short)0, acc, false, false);
    }
    __syncthreads();
  }
  if (!active) return;

  const int col = N0 + r16;
  const int rbase = M0 + ((lane < 16) ? 0 : 8);
  #pragma unroll
  for (int i = 0; i < 8; ++i) {
    int row = rbase + i;
    if (row >= Mact) continue;
    float v = acc[i];
    if (bias) v += bias[(long)k * biasStrideK + row];
    if (act == 1) v = softplusf(v);
    else if (act == 2) v = geluf(v);
    if (res) v += res[(long)s * resStride + (long)row * N + col];
    long o = (long)row * N + col;
    if (C)   C[(long)s * cStride + o] = v;
    if (C16) C16[(long)s * c16Stride + o] = (_Float16)v;
  }
}

// ------------------------------------------------------------------
// Depthwise 3x3 conv (pad 1) + bias + SiLU on rows 0..191 of xz
// ------------------------------------------------------------------
__global__ void dwconv_silu(const float* __restrict__ xz, const float* __restrict__ cw,
                            const float* __restrict__ cb, float* __restrict__ xc)
{
  long idx = (long)blockIdx.x * 256 + threadIdx.x;   // over BATCH*DINN*LL
  if (idx >= (long)BATCH * DINN * LL) return;
  int l = (int)(idx % LL);
  int d = (int)((idx / LL) % DINN);
  int b = (int)(idx / ((long)DINN * LL));
  int h = l / WW, w = l % WW;
  const float* src = xz + ((long)b * (2 * DINN) + d) * LL;
  const float* wk = cw + d * 9;
  float acc = cb[d];
  #pragma unroll
  for (int i = 0; i < 3; ++i) {
    int hy = h + i - 1;
    if (hy < 0 || hy >= HH) continue;
    #pragma unroll
    for (int j = 0; j < 3; ++j) {
      int wx = w + j - 1;
      if (wx < 0 || wx >= WW) continue;
      acc += wk[i * 3 + j] * src[hy * WW + wx];
    }
  }
  xc[idx] = siluf(acc);
}

// ------------------------------------------------------------------
// CrossScan: build 4 direction sequences, f32 -> f16
// ------------------------------------------------------------------
__global__ void cross_scan(const float* __restrict__ xc, _Float16* __restrict__ xs16)
{
  long idx = (long)blockIdx.x * 256 + threadIdx.x;   // over BATCH*DINN*LL
  if (idx >= (long)BATCH * DINN * LL) return;
  int l = (int)(idx % LL);
  int d = (int)((idx / LL) % DINN);
  int b = (int)(idx / ((long)DINN * LL));
  const float* src = xc + ((long)b * DINN + d) * LL;
  _Float16* dst = xs16 + (((long)b * KD) * DINN + d) * LL;
  const long ks = (long)DINN * LL;
  int l1 = (l % HH) * WW + (l / HH);
  int lr = LL - 1 - l;
  int l3 = (lr % HH) * WW + (lr / HH);
  dst[l]          = (_Float16)src[l];
  dst[ks + l]     = (_Float16)src[l1];
  dst[2 * ks + l] = (_Float16)src[lr];
  dst[3 * ks + l] = (_Float16)src[l3];
}

// ------------------------------------------------------------------
// Build padded dt B-matrix: rows 0..5 from x_dbl, rows 6..31 zero
// ------------------------------------------------------------------
__global__ void dts_prep(const float* __restrict__ x_dbl, _Float16* __restrict__ dts16)
{
  long idx = (long)blockIdx.x * 256 + threadIdx.x;   // over BATCH*KD*32*LL
  if (idx >= (long)BATCH * KD * DT_KPAD * LL) return;
  int l = (int)(idx % LL);
  int r = (int)((idx / LL) % DT_KPAD);
  int s = (int)(idx / ((long)DT_KPAD * LL));
  float v = (r < RTK) ? x_dbl[((long)s * XPR + r) * LL + l] : 0.f;
  dts16[idx] = (_Float16)v;
}

// ------------------------------------------------------------------
// Selective scan: one block per (b,k), one thread per channel d,
// 16-entry diagonal state in registers. B/C rows for each 8-step chunk
// are transpose-gathered into LDS with global_load_async_to_lds_b32
// (per-lane global src -> per-lane LDS dst, tracked by ASYNCcnt).
// ------------------------------------------------------------------
__global__ void __launch_bounds__(DINN) selective_scan(
    const float* __restrict__ delta, const _Float16* __restrict__ xs16,
    const float* __restrict__ x_dbl, const float* __restrict__ A_logs,
    const float* __restrict__ Ds, float* __restrict__ ys)
{
  const int s = blockIdx.x;            // b*KD + k
  const int k = s % KD;
  const int d = threadIdx.x;           // 0..191
  const float* Arow = A_logs + ((long)k * DINN + d) * NST;
  float a[NST];
  #pragma unroll
  for (int n = 0; n < NST; ++n) a[n] = -__expf(Arow[n]);
  const float Dv = Ds[k * DINN + d];
  const float*    dp = delta + ((long)s * DINN + d) * LL;
  const _Float16* up = xs16  + ((long)s * DINN + d) * LL;
  const float*    Bp = x_dbl + ((long)s * XPR + RTK) * LL;
  const float*    Cp = x_dbl + ((long)s * XPR + RTK + NST) * LL;
  float*          yp = ys    + ((long)s * DINN + d) * LL;

  float h[NST];
  #pragma unroll
  for (int n = 0; n < NST; ++n) h[n] = 0.f;

  __shared__ float sB[8][NST];
  __shared__ float sC[8][NST];

  for (int l0 = 0; l0 < LL; l0 += 8) {
    // async transpose-gather of the B/C chunk into LDS
    for (int i = threadIdx.x; i < 8 * 2 * NST; i += DINN) {
      int cc = i >> 5, n = i & 31;
      const float* g;
      float* t;
      if (n < NST) { g = Bp + (long)n * LL + l0 + cc;        t = &sB[cc][n]; }
      else         { g = Cp + (long)(n - NST) * LL + l0 + cc; t = &sC[cc][n - NST]; }
      unsigned ldsoff = (unsigned)(size_t)t;                 // LDS offset (low 32b)
      unsigned long long gaddr = (unsigned long long)(size_t)g;
      asm volatile("global_load_async_to_lds_b32 %0, %1, off"
                   :: "v"(ldsoff), "v"(gaddr) : "memory");
    }
    asm volatile("s_wait_asynccnt 0x0" ::: "memory");
    __syncthreads();
    #pragma unroll
    for (int cc = 0; cc < 8; ++cc) {
      int l = l0 + cc;
      float dt = dp[l];
      float u  = (float)up[l];
      float du = dt * u;
      float y = 0.f;
      #pragma unroll
      for (int n = 0; n < NST; ++n) {
        float hn = h[n] * __expf(dt * a[n]) + du * sB[cc][n];
        h[n] = hn;
        y += hn * sC[cc][n];
      }
      yp[l] = y + Dv * u;
    }
    __syncthreads();
  }
}

// ------------------------------------------------------------------
// CrossMerge: undo flips / transpose and sum 4 directions
// ------------------------------------------------------------------
__global__ void cross_merge(const float* __restrict__ ys, float* __restrict__ ym)
{
  long idx = (long)blockIdx.x * 256 + threadIdx.x;   // over BATCH*DINN*LL
  if (idx >= (long)BATCH * DINN * LL) return;
  int l = (int)(idx % LL);
  int d = (int)((idx / LL) % DINN);
  int b = (int)(idx / ((long)DINN * LL));
  const long ks = (long)DINN * LL;
  const float* base = ys + (((long)b * KD) * DINN + d) * LL;
  int lp = (l % WW) * HH + (l / WW);
  float v = base[l] + base[2 * ks + (LL - 1 - l)]
          + base[ks + lp] + base[3 * ks + (LL - 1 - lp)];
  ym[((long)b * DINN + d) * LL + l] = v;
}

// ------------------------------------------------------------------
// out_norm LN over DINN + multiply silu(z) gate, write f16
// ------------------------------------------------------------------
__global__ void outnorm_zmul_f16(const float* __restrict__ ym, const float* __restrict__ xz,
                                 const float* __restrict__ w, const float* __restrict__ b,
                                 _Float16* __restrict__ y16)
{
  int p = blockIdx.x * 256 + threadIdx.x;
  if (p >= BATCH * LL) return;
  int l = p % LL, bb = p / LL;
  const float* base = ym + (long)bb * DINN * LL + l;
  float s = 0.f, s2 = 0.f;
  for (int c = 0; c < DINN; ++c) { float v = base[(long)c * LL]; s += v; s2 += v * v; }
  float mu = s / (float)DINN;
  float rs = rsqrtf(s2 / (float)DINN - mu * mu + 1e-5f);
  const float* zb = xz + ((long)bb * (2 * DINN) + DINN) * LL + l;
  _Float16* ob = y16 + (long)bb * DINN * LL + l;
  for (int c = 0; c < DINN; ++c) {
    float t = (base[(long)c * LL] - mu) * rs * w[c] + b[c];
    float z = siluf(zb[(long)c * LL]);
    ob[(long)c * LL] = (_Float16)(t * z);
  }
}

// ==================================================================
extern "C" void kernel_launch(void* const* d_in, const int* in_sizes, int n_in,
                              void* d_out, int out_size, void* d_ws, size_t ws_size,
                              hipStream_t stream)
{
  (void)in_sizes; (void)n_in; (void)out_size; (void)ws_size;
  const float* x          = (const float*)d_in[0];
  const float* norm1_w    = (const float*)d_in[1];
  const float* norm1_b    = (const float*)d_in[2];
  const float* in_proj_w  = (const float*)d_in[3];
  const float* conv_w     = (const float*)d_in[4];
  const float* conv_b     = (const float*)d_in[5];
  const float* x_proj_w   = (const float*)d_in[6];
  const float* dt_projs_w = (const float*)d_in[7];
  const float* dt_projs_b = (const float*)d_in[8];
  const float* A_logs     = (const float*)d_in[9];
  const float* Ds         = (const float*)d_in[10];
  const float* out_norm_w = (const float*)d_in[11];
  const float* out_norm_b = (const float*)d_in[12];
  const float* out_proj_w = (const float*)d_in[13];
  const float* norm2_w    = (const float*)d_in[14];
  const float* norm2_b    = (const float*)d_in[15];
  const float* fc1_w      = (const float*)d_in[16];
  const float* fc1_b      = (const float*)d_in[17];
  const float* fc2_w      = (const float*)d_in[18];
  const float* fc2_b      = (const float*)d_in[19];
  float* out = (float*)d_out;

  char* ws = (char*)d_ws;
  size_t off = 0;
  auto alloc = [&](size_t bytes) -> void* {
    void* p = ws + off;
    off = (off + bytes + 255) & ~(size_t)255;
    return p;
  };

  _Float16* h16    = (_Float16*)alloc((size_t)BATCH * CDIM * LL * 2);
  _Float16* w_in   = (_Float16*)alloc((size_t)384 * 96 * 2);
  _Float16* w_xp   = (_Float16*)alloc((size_t)KD * XPR_MPAD * DINN * 2);
  _Float16* w_dt   = (_Float16*)alloc((size_t)KD * DINN * DT_KPAD * 2);
  _Float16* w_out  = (_Float16*)alloc((size_t)CDIM * DINN * 2);
  _Float16* w_fc1  = (_Float16*)alloc((size_t)(4 * CDIM) * CDIM * 2);
  _Float16* w_fc2  = (_Float16*)alloc((size_t)CDIM * (4 * CDIM) * 2);
  float*    xz     = (float*)alloc((size_t)BATCH * 2 * DINN * LL * 4);
  float*    xc     = (float*)alloc((size_t)BATCH * DINN * LL * 4);   // reused as ymerge
  _Float16* xs16   = (_Float16*)alloc((size_t)BATCH * KD * DINN * LL * 2);
  float*    x_dbl  = (float*)alloc((size_t)BATCH * KD * XPR * LL * 4);
  _Float16* dts16  = (_Float16*)alloc((size_t)BATCH * KD * DT_KPAD * LL * 2);
  float*    delta  = (float*)alloc((size_t)BATCH * KD * DINN * LL * 4);
  float*    ys     = (float*)alloc((size_t)BATCH * KD * DINN * LL * 4); // reused as g1_16
  _Float16* y16    = (_Float16*)alloc((size_t)BATCH * DINN * LL * 2);
  _Float16* hm16   = (_Float16*)alloc((size_t)BATCH * CDIM * LL * 2);
  float*    xres   = (float*)alloc((size_t)BATCH * CDIM * LL * 4);
  float*    ymerge = xc;
  _Float16* g1_16  = (_Float16*)ys;

  auto gemm = [&](int Kpad, const _Float16* A, int aStrideK, const _Float16* Bm, long bStride,
                  float* C, long cStride, _Float16* C16, long c16Stride,
                  const float* bias, int biasStrideK, const float* res, long resStride,
                  int M, int Mact, int nK, int act) {
    int mgroups = ((M >> 4) + 7) >> 3;
    dim3 g(mgroups * (LL >> 4), BATCH * nK);
    switch (Kpad) {
      case 32:  wmma_gemm<32><<<g,256,0,stream>>>(A,aStrideK,Bm,bStride,C,cStride,C16,c16Stride,bias,biasStrideK,res,resStride,M,Mact,LL,nK,act,mgroups); break;
      case 96:  wmma_gemm<96><<<g,256,0,stream>>>(A,aStrideK,Bm,bStride,C,cStride,C16,c16Stride,bias,biasStrideK,res,resStride,M,Mact,LL,nK,act,mgroups); break;
      case 192: wmma_gemm<192><<<g,256,0,stream>>>(A,aStrideK,Bm,bStride,C,cStride,C16,c16Stride,bias,biasStrideK,res,resStride,M,Mact,LL,nK,act,mgroups); break;
      default:  wmma_gemm<384><<<g,256,0,stream>>>(A,aStrideK,Bm,bStride,C,cStride,C16,c16Stride,bias,biasStrideK,res,resStride,M,Mact,LL,nK,act,mgroups); break;
    }
  };

  // --- pack weights to padded f16 ---
  pack_w16<<<(384 * 96 + 255) / 256, 256, 0, stream>>>(in_proj_w, w_in, 384, 96, 384, 96, 1);
  pack_w16<<<(KD * XPR_MPAD * DINN + 255) / 256, 256, 0, stream>>>(x_proj_w, w_xp, XPR, DINN, XPR_MPAD, DINN, KD);
  pack_w16<<<(KD * DINN * DT_KPAD + 255) / 256, 256, 0, stream>>>(dt_projs_w, w_dt, DINN, RTK, DINN, DT_KPAD, KD);
  pack_w16<<<(CDIM * DINN + 255) / 256, 256, 0, stream>>>(out_proj_w, w_out, CDIM, DINN, CDIM, DINN, 1);
  pack_w16<<<(4 * CDIM * CDIM + 255) / 256, 256, 0, stream>>>(fc1_w, w_fc1, 4 * CDIM, CDIM, 4 * CDIM, CDIM, 1);
  pack_w16<<<(CDIM * 4 * CDIM + 255) / 256, 256, 0, stream>>>(fc2_w, w_fc2, CDIM, 4 * CDIM, CDIM, 4 * CDIM, 1);

  const int pixBlocks = (BATCH * LL + 255) / 256;
  const int elBlocksD = (int)(((long)BATCH * DINN * LL + 255) / 256);

  // --- SS2D branch ---
  ln2d_f16<<<pixBlocks, 256, 0, stream>>>(x, norm1_w, norm1_b, h16, CDIM);
  gemm(CDIM, w_in, 0, h16, (long)CDIM * LL, xz, (long)2 * DINN * LL, nullptr, 0,
       nullptr, 0, nullptr, 0, 2 * DINN, 2 * DINN, 1, 0);
  dwconv_silu<<<elBlocksD, 256, 0, stream>>>(xz, conv_w, conv_b, xc);
  cross_scan<<<elBlocksD, 256, 0, stream>>>(xc, xs16);
  gemm(DINN, w_xp, XPR_MPAD * DINN, xs16, (long)DINN * LL, x_dbl, (long)XPR * LL, nullptr, 0,
       nullptr, 0, nullptr, 0, XPR_MPAD, XPR, KD, 0);
  dts_prep<<<(int)(((long)BATCH * KD * DT_KPAD * LL + 255) / 256), 256, 0, stream>>>(x_dbl, dts16);
  gemm(DT_KPAD, w_dt, DINN * DT_KPAD, dts16, (long)DT_KPAD * LL, delta, (long)DINN * LL, nullptr, 0,
       dt_projs_b, DINN, nullptr, 0, DINN, DINN, KD, 1 /*softplus*/);
  selective_scan<<<BATCH * KD, DINN, 0, stream>>>(delta, xs16, x_dbl, A_logs, Ds, ys);
  cross_merge<<<elBlocksD, 256, 0, stream>>>(ys, ymerge);
  outnorm_zmul_f16<<<pixBlocks, 256, 0, stream>>>(ymerge, xz, out_norm_w, out_norm_b, y16);
  gemm(DINN, w_out, 0, y16, (long)DINN * LL, xres, (long)CDIM * LL, nullptr, 0,
       nullptr, 0, x, (long)CDIM * LL, CDIM, CDIM, 1, 0);   // + residual x

  // --- MLP branch ---
  ln2d_f16<<<pixBlocks, 256, 0, stream>>>(xres, norm2_w, norm2_b, hm16, CDIM);
  gemm(CDIM, w_fc1, 0, hm16, (long)CDIM * LL, nullptr, 0, g1_16, (long)4 * CDIM * LL,
       fc1_b, 0, nullptr, 0, 4 * CDIM, 4 * CDIM, 1, 2 /*gelu*/);
  gemm(4 * CDIM, w_fc2, 0, g1_16, (long)4 * CDIM * LL, out, (long)CDIM * LL, nullptr, 0,
       fc2_b, 0, xres, (long)CDIM * LL, CDIM, CDIM, 1, 0); // + residual
}
